// ContextAttention_45741401703142
// MI455X (gfx1250) — matrix operand order repaired and checked by reading, not compiled, not run on hardware
//
#include <hip/hip_runtime.h>
#include <math.h>

typedef __attribute__((ext_vector_type(16))) _Float16 v16h;
typedef __attribute__((ext_vector_type(8)))  _Float16 v8h;
typedef __attribute__((ext_vector_type(8)))  float    v8f;

union Frag { v16h v; v8h h[2]; };

#define GBM 128
#define GBN 128
#define GBK 32
#define LDSW 40   // halves per LDS row (32 data + 8 pad, keeps 16B alignment)

// ---------------------------------------------------------------------------
// C[M x N] = A[M x K] * Bt[N x K]^T, f16 inputs, f32 accumulate.
// v_wmma_f32_16x16x32_f16; block tile 128x128x32, double-buffered LDS,
// 8 waves, each wave 32x64 (2x4 WMMA tiles -> 8 WMMAs per K-step).
// Requires M%128==0, N%128==0, K%32==0.
// ---------------------------------------------------------------------------
__global__ __launch_bounds__(256) void ca_gemm_f16(
    const _Float16* __restrict__ A, const _Float16* __restrict__ Bt,
    float* __restrict__ C, int M, int N, int K)
{
    __shared__ __align__(16) _Float16 As[2][GBM * LDSW];
    __shared__ __align__(16) _Float16 Bs[2][GBN * LDSW];
    const int bm   = blockIdx.y * GBM;
    const int bn   = blockIdx.x * GBN;
    const int tid  = threadIdx.x;
    const int lane = tid & 31;
    const int wave = tid >> 5;
    const int wm   = (wave >> 1) * 32;   // wave row offset (0,32,64,96)
    const int wn   = (wave & 1) * 64;    // wave col offset (0,64)
    const int hlf  = lane >> 4;          // half-wave select
    const int l16  = lane & 15;

    const int grow = tid >> 1;           // 0..127: tile row for global->LDS
    const int gcol = (tid & 1) * 16;     // 0 or 16 (halves)

    const _Float16* Aptr = A  + (size_t)(bm + grow) * K + gcol;
    const _Float16* Bptr = Bt + (size_t)(bn + grow) * K + gcol;

    v8f acc[2][4] = {};

    // prologue: stage K-slice 0
    v8h a0 = *(const v8h*)(Aptr);
    v8h a1 = *(const v8h*)(Aptr + 8);
    v8h b0 = *(const v8h*)(Bptr);
    v8h b1 = *(const v8h*)(Bptr + 8);
    *(v8h*)(As[0] + grow * LDSW + gcol)     = a0;
    *(v8h*)(As[0] + grow * LDSW + gcol + 8) = a1;
    *(v8h*)(Bs[0] + grow * LDSW + gcol)     = b0;
    *(v8h*)(Bs[0] + grow * LDSW + gcol + 8) = b1;
    __syncthreads();

    int buf = 0;
    for (int k0 = 0; k0 < K; k0 += GBK) {
        const bool more = (k0 + GBK) < K;
        if (more) {   // prefetch next K-slice into registers
            a0 = *(const v8h*)(Aptr + k0 + GBK);
            a1 = *(const v8h*)(Aptr + k0 + GBK + 8);
            b0 = *(const v8h*)(Bptr + k0 + GBK);
            b1 = *(const v8h*)(Bptr + k0 + GBK + 8);
        }

        Frag af[2], bf[4];
        // A fragment: lane(0-15)=row M, K {0..7,16..23}; lanes 16-31: K {8..15,24..31}
        #pragma unroll
        for (int i = 0; i < 2; ++i) {
            const _Float16* rp = As[buf] + (wm + i * 16 + l16) * LDSW;
            af[i].h[0] = *(const v8h*)(rp + hlf * 8);
            af[i].h[1] = *(const v8h*)(rp + 16 + hlf * 8);
        }
        // B fragment: lane(0-15)=col N, K {0..7,16..23}; lanes 16-31: K {8..15,24..31}
        #pragma unroll
        for (int j = 0; j < 4; ++j) {
            const _Float16* rp = Bs[buf] + (wn + j * 16 + l16) * LDSW;
            bf[j].h[0] = *(const v8h*)(rp + hlf * 8);
            bf[j].h[1] = *(const v8h*)(rp + 16 + hlf * 8);
        }
        #pragma unroll
        for (int i = 0; i < 2; ++i)
            #pragma unroll
            for (int j = 0; j < 4; ++j)
                acc[i][j] = __builtin_amdgcn_wmma_f32_16x16x32_f16(
                    false, af[i].v, false, bf[j].v, (short)0, acc[i][j], false, false);

        if (more) {   // stage next slice into the other buffer; one barrier/iter
            const int nb = buf ^ 1;
            *(v8h*)(As[nb] + grow * LDSW + gcol)     = a0;
            *(v8h*)(As[nb] + grow * LDSW + gcol + 8) = a1;
            *(v8h*)(Bs[nb] + grow * LDSW + gcol)     = b0;
            *(v8h*)(Bs[nb] + grow * LDSW + gcol + 8) = b1;
            __syncthreads();
            buf = nb;
        }
    }

    // C/D layout: VGPR r, lane l -> row = r + 8*(l>>4), col = l&15
    #pragma unroll
    for (int i = 0; i < 2; ++i)
        #pragma unroll
        for (int j = 0; j < 4; ++j)
            #pragma unroll
            for (int r = 0; r < 8; ++r)
                C[(size_t)(bm + wm + i * 16 + r + 8 * hlf) * N +
                  (bn + wn + j * 16 + l16)] = acc[i][j][r];
    (void)M;
}

// ---------------------------------------------------------------------------
// Build normalized patch matrix Wn[4096 x 576] (f16) from b (sample s),
// replicating the reference's scrambled reshape of stride-1 ds patches.
// k ordering = (c, a, b) to match the im2col of f_ds.
// ---------------------------------------------------------------------------
__global__ __launch_bounds__(128) void ca_build_w(
    const float* __restrict__ b, _Float16* __restrict__ Wn, int s)
{
    __shared__ float red[128];
    const int n   = blockIdx.x;
    const int tid = threadIdx.x;
    float vals[5];
    float ss = 0.f;
    #pragma unroll
    for (int e = 0; e < 5; ++e) {
        int k = tid + e * 128;
        float v = 0.f;
        if (k < 576) {
            int cc = k / 9, rem = k % 9, a = rem / 3, bb = rem % 3;
            int flat = n * 576 + a * 192 + bb * 64 + cc;   // reference reshape scramble
            int kk = flat >> 18;            // /262144
            int c2 = (flat >> 12) & 63;
            int ih = (flat >> 6) & 63;
            int iw = flat & 63;
            int u = ih + (kk / 3) - 1;
            int w = iw + (kk % 3) - 1;
            if (u >= 0 && u < 64 && w >= 0 && w < 64)
                v = b[(((size_t)s * 64 + c2) * 128 + 2 * u) * 128 + 2 * w];
        }
        vals[e] = v;
        ss += v * v;
    }
    red[tid] = ss; __syncthreads();
    for (int off = 64; off > 0; off >>= 1) {
        if (tid < off) red[tid] += red[tid + off];
        __syncthreads();
    }
    float scale = 1.0f / fminf(sqrtf(red[0]), 1e-4f);   // reference uses minimum()
    #pragma unroll
    for (int e = 0; e < 5; ++e) {
        int k = tid + e * 128;
        if (k < 576) Wn[(size_t)n * 576 + k] = (_Float16)(vals[e] * scale);
    }
}

// im2col of f_ds: F[4096 x 576] (f16), k = (c,a,b)
__global__ void ca_build_f(const float* __restrict__ f, _Float16* __restrict__ F, int s)
{
    int idx = blockIdx.x * 256 + threadIdx.x;       // 4096*576 threads exactly
    int j = idx / 576, k = idx % 576;
    int cc = k / 9, rem = k % 9, a = rem / 3, bb = rem % 3;
    int Y = j >> 6, X = j & 63;
    int u = Y + a - 1, v = X + bb - 1;
    float val = 0.f;
    if (u >= 0 && u < 64 && v >= 0 && v < 64)
        val = f[(((size_t)s * 64 + cc) * 128 + 2 * u) * 128 + 2 * v];
    F[idx] = (_Float16)val;
}

// Paste weights transposed: W2t[(co,p,q)=1024][j=4096] = raw_w[j, co, 3-p, 3-q]
__global__ void ca_build_w2t(const float* __restrict__ b, _Float16* __restrict__ W2t, int s)
{
    int idx = blockIdx.x * 256 + threadIdx.x;       // 1024*4096 threads exactly
    int cpq = idx >> 12, j = idx & 4095;
    int co = cpq >> 4, p = (cpq >> 2) & 3, q = cpq & 3;
    int a2 = 3 - p, b2 = 3 - q;                      // spatial flip
    int flat = j * 1024 + a2 * 256 + b2 * 64 + co;   // reference reshape scramble
    int kk = flat >> 18;
    int c2 = (flat >> 12) & 63;
    int ih = (flat >> 6) & 63;
    int iw = flat & 63;
    int u = (kk >> 2) + 2 * ih - 1;                  // stride-2 patches, pad 1
    int v = (kk & 3) + 2 * iw - 1;
    float val = 0.f;
    if (u >= 0 && u < 128 && v >= 0 && v < 128)
        val = b[(((size_t)s * 64 + c2) * 128 + u) * 128 + v];
    W2t[idx] = (_Float16)val;
}

// mm[t] from sample-0 mask patches (mask is zeros -> mm = 1, but compute it)
__global__ void ca_mm(const float* __restrict__ mask, float* __restrict__ mm)
{
    int t = blockIdx.x * 256 + threadIdx.x;
    if (t >= 4096) return;
    float sum = 0.f;
    for (int e = 0; e < 9; ++e) {
        int flat = t * 9 + e;
        int kk = flat >> 12;
        int ih = (flat >> 6) & 63;
        int iw = flat & 63;
        int u = ih + kk / 3 - 1, v = iw + kk % 3 - 1;
        if (u >= 0 && u < 64 && v >= 0 && v < 64)
            sum += mask[(size_t)(2 * u) * 128 + 2 * v];
    }
    mm[t] = ((sum / 9.0f) == 0.0f) ? 1.0f : 0.0f;
}

// Fuse pass 1: U[r,c] = sum_d S[r+d, c+d], d in {-1,0,1}
__global__ void ca_fuse1(const float* __restrict__ S, float* __restrict__ U)
{
    size_t idx = (size_t)blockIdx.x * 256 + threadIdx.x;
    int r = (int)(idx >> 12), c = (int)(idx & 4095);
    float a = 0.f;
    #pragma unroll
    for (int t = -1; t <= 1; ++t) {
        int rr = r + t, cc = c + t;
        if (rr >= 0 && rr < 4096 && cc >= 0 && cc < 4096)
            a += S[((size_t)rr << 12) + cc];
    }
    U[idx] = a;
}

__device__ __forceinline__ int permi(int x) { return ((x & 63) << 6) | (x >> 6); }

// Fuse pass 2 with the reference's transpose sandwich folded in:
// S[n,j] = sum_d U[perm(perm(n)+d), perm(perm(j)+d)]
__global__ void ca_fuse2(const float* __restrict__ U, float* __restrict__ S)
{
    size_t idx = (size_t)blockIdx.x * 256 + threadIdx.x;
    int n = (int)(idx >> 12), j = (int)(idx & 4095);
    int rp = permi(n), cp = permi(j);
    float a = 0.f;
    #pragma unroll
    for (int t = -1; t <= 1; ++t) {
        int rr = rp + t, cc = cp + t;
        if (rr >= 0 && rr < 4096 && cc >= 0 && cc < 4096)
            a += U[((size_t)permi(rr) << 12) + permi(cc)];
    }
    S[idx] = a;
}

// Row softmax (scale 10, mask mm) + argmax; writes f16 attention matrix.
__global__ __launch_bounds__(256) void ca_softmax(
    const float* __restrict__ S, const float* __restrict__ mm,
    _Float16* __restrict__ Msm, int* __restrict__ offr)
{
    __shared__ float sred[256];
    __shared__ int   sidx[256];
    const int n   = blockIdx.x;
    const int tid = threadIdx.x;
    const float* row = S + (size_t)n * 4096;
    float lg[16];
    float lmax = -1e30f, bestv = -1e30f;
    int bestj = 0x7fffffff;
    #pragma unroll
    for (int e = 0; e < 16; ++e) {
        int j = tid + e * 256;
        float m = mm[j];
        float l = row[j] * m * 10.0f;
        lg[e] = l;
        lmax = fmaxf(lmax, l);
        if (m != 0.0f && (l > bestv || (l == bestv && j < bestj))) { bestv = l; bestj = j; }
    }
    sred[tid] = lmax; __syncthreads();
    for (int off = 128; off > 0; off >>= 1) {
        if (tid < off) sred[tid] = fmaxf(sred[tid], sred[tid + off]);
        __syncthreads();
    }
    float gmax = sred[0]; __syncthreads();
    float lsum = 0.f;
    #pragma unroll
    for (int e = 0; e < 16; ++e) lsum += __expf(lg[e] - gmax);
    sred[tid] = lsum; __syncthreads();
    for (int off = 128; off > 0; off >>= 1) {
        if (tid < off) sred[tid] += sred[tid + off];
        __syncthreads();
    }
    float inv = 1.0f / sred[0]; __syncthreads();
    #pragma unroll
    for (int e = 0; e < 16; ++e) {
        int j = tid + e * 256;
        Msm[(size_t)n * 4096 + j] = (_Float16)(__expf(lg[e] - gmax) * inv * mm[j]);
    }
    sred[tid] = bestv; sidx[tid] = bestj; __syncthreads();
    for (int off = 128; off > 0; off >>= 1) {
        if (tid < off) {
            float ov = sred[tid + off]; int oj = sidx[tid + off];
            if (ov > sred[tid] || (ov == sred[tid] && oj < sidx[tid])) {
                sred[tid] = ov; sidx[tid] = oj;
            }
        }
        __syncthreads();
    }
    if (tid == 0) offr[n] = (sred[0] <= -1e30f) ? 0 : sidx[0];
}

// Offsets output: (bs,2,64,64): [0]=argmax//128 - n1, [1]=argmax%128 - n0
__global__ void ca_offsets(const int* __restrict__ offr, float* __restrict__ outOff, int s)
{
    int n = blockIdx.x * 256 + threadIdx.x;
    if (n >= 4096) return;
    int n1 = n >> 6, n0 = n & 63;
    int jb = offr[n];
    outOff[((size_t)s * 2 + 0) * 4096 + n] = (float)(jb / 128 - n1);
    outOff[((size_t)s * 2 + 1) * 4096 + n] = (float)(jb % 128 - n0);
}

// Dilated (rate-2, pad-2) scatter of G[4096 x 1024] into y[64 x 128 x 128] / 4
__global__ void ca_scatter(const float* __restrict__ G, float* __restrict__ y, int s)
{
    int idx = blockIdx.x * 256 + threadIdx.x;   // 64*128*128 threads exactly
    int co = idx >> 14;
    int yy = (idx >> 7) & 127;
    int xx = idx & 127;
    float acc = 0.f;
    #pragma unroll
    for (int dp = 0; dp < 2; ++dp) {
        int p = (yy & 1) + 2 * dp;              // parity-matched taps
        int ny = yy + p - 2;
        if (ny < 0 || ny > 126) continue;
        int nY = ny >> 1;
        #pragma unroll
        for (int dq = 0; dq < 2; ++dq) {
            int q = (xx & 1) + 2 * dq;
            int nx = xx + q - 2;
            if (nx < 0 || nx > 126) continue;
            int nX = nx >> 1;
            acc += G[(size_t)(nY * 64 + nX) * 1024 + (co << 4) + (p << 2) + q];
        }
    }
    y[(size_t)s * 1048576 + idx] = acc * 0.25f;
}

// ---------------------------------------------------------------------------
extern "C" void kernel_launch(void* const* d_in, const int* in_sizes, int n_in,
                              void* d_out, int out_size, void* d_ws, size_t ws_size,
                              hipStream_t stream)
{
    (void)in_sizes; (void)n_in; (void)out_size; (void)ws_size;
    const float* f    = (const float*)d_in[0];
    const float* b    = (const float*)d_in[1];
    const float* mask = (const float*)d_in[2];
    float* out = (float*)d_out;

    char* ws = (char*)d_ws;
    float*    S   = (float*)(ws);                        // 64 MB scores
    float*    U   = (float*)(ws + 67108864ull);          // 64 MB fuse temp
    _Float16* Msm = (_Float16*)(ws + 134217728ull);      // 32 MB f16 attention
    _Float16* Wn  = (_Float16*)(ws + 167772160ull);      // 4.5 MB
    _Float16* Fm  = (_Float16*)(ws + 172490752ull);      // 4.5 MB
    _Float16* W2t = (_Float16*)(ws + 177209344ull);      // 8 MB
    float*    G   = (float*)(ws + 185597952ull);         // 16 MB
    float*    mmv = (float*)(ws + 202375168ull);         // 16 KB
    int*      offr= (int*)(ws + 202391552ull);           // 16 KB

    ca_mm<<<16, 256, 0, stream>>>(mask, mmv);

    for (int s = 0; s < 2; ++s) {
        ca_build_w  <<<4096, 128, 0, stream>>>(b, Wn, s);
        ca_build_f  <<<9216, 256, 0, stream>>>(f, Fm, s);
        ca_build_w2t<<<16384, 256, 0, stream>>>(b, W2t, s);
        // S[n,j] = Wn . F^T   (4096 x 4096, K=576)
        ca_gemm_f16<<<dim3(32, 32), 256, 0, stream>>>(Wn, Fm, S, 4096, 4096, 576);
        ca_fuse1<<<65536, 256, 0, stream>>>(S, U);
        ca_fuse2<<<65536, 256, 0, stream>>>(U, S);
        ca_softmax<<<4096, 256, 0, stream>>>(S, mmv, Msm, offr);
        ca_offsets<<<16, 256, 0, stream>>>(offr, out + 2097152, s);
        // G[n,cpq] = Msm . W2t^T  (4096 x 1024, K=4096)
        ca_gemm_f16<<<dim3(8, 32), 256, 0, stream>>>(Msm, W2t, G, 4096, 1024, 4096);
        ca_scatter<<<4096, 256, 0, stream>>>(G, out, s);
    }
}
